// EfficientSparseCrossAttention_89026082112156
// MI455X (gfx1250) — matrix operand order repaired and checked by reading, not compiled, not run on hardware
//
#include <hip/hip_runtime.h>
#include <hip/hip_bf16.h>

// ---------------- problem constants (from reference) ----------------
#define Bn  16
#define Sn  1024
#define Pn  1024
#define Dn  128        // model dim (= K of all GEMMs)
#define Hn  8          // heads
#define HDn 16         // head dim
#define MSn 32         // MLP hidden
#define En  524288     // edges
#define TANH_CLIP 10.0f
#define EPSc 1e-9f

typedef __attribute__((ext_vector_type(16))) _Float16 v16h;
typedef __attribute__((ext_vector_type(8)))  float    v8f;

// =====================================================================
// WMMA GEMM: C[M,N] = A[M,K=128] * W[128,N], fp32 in/out, f16 WMMA core.
//
// Block = 256 threads (8 waves). Block tile = 128 rows x 64 cols.
//  - W slab (128 x 64) is converted to f16 once and staged in LDS,
//    transposed to [n][k] so each B fragment is one contiguous 32B
//    ds_load per lane.
//  - Each wave owns 16 rows x 64 cols = 4 accumulators; the A fragment
//    is loaded once per K-step (4x global_load_b128) and reused by 4
//    v_wmma_f32_16x16x32_f16 ops. K=128 fully unrolled -> 16 WMMA/wave.
//  - Grid sized exactly: EXEC is all-ones in every WMMA wave.
// =====================================================================
__global__ __launch_bounds__(256)
void gemm_wmma_f16(const float* __restrict__ A,
                   const float* __restrict__ W,
                   float* __restrict__ C,
                   int M, int N) {
  __shared__ _Float16 wlds[64 * Dn];          // [n][k], 16 KB

  const int nslabs = N >> 6;                  // N/64 slabs
  const int mb     = blockIdx.x / nslabs;     // 128-row block
  const int ns     = blockIdx.x % nslabs;     // 64-col slab
  const int tid    = threadIdx.x;

  // ---- stage W slab into LDS as f16, transposed to [n][k] ----
  // global W is row-major (k, n): element (k, nslab*64+n)
#pragma unroll
  for (int i = 0; i < (Dn * 64) / 256; ++i) { // 32 iters
    int idx = tid + i * 256;
    int k = idx >> 6;                         // /64
    int n = idx & 63;
    wlds[n * Dn + k] = (_Float16)W[(size_t)k * N + ns * 64 + n];
  }
  __syncthreads();

  const int wave = tid >> 5;
  const int lane = tid & 31;
  const int l16  = lane & 15;
  const int hs   = lane >> 4;                 // lane half

  const int row  = mb * 128 + wave * 16 + l16;      // A row for this lane
  const float4* __restrict__ a4 = (const float4*)(A + (size_t)row * Dn);

  v8f acc0 = {}, acc1 = {}, acc2 = {}, acc3 = {};

#pragma unroll
  for (int k0 = 0; k0 < Dn; k0 += 32) {
    // ---- A fragment (16x32 f16): K = ka..ka+15
    const int ka = k0 + hs * 16;
    const int i0 = ka >> 2;
    float4 p0 = a4[i0], p1 = a4[i0 + 1];      // K = ka+0..7
    float4 p2 = a4[i0 + 2], p3 = a4[i0 + 3];  // K = ka+8..15
    v16h a;
    a[0]=(_Float16)p0.x; a[1]=(_Float16)p0.y; a[2]=(_Float16)p0.z; a[3]=(_Float16)p0.w;
    a[4]=(_Float16)p1.x; a[5]=(_Float16)p1.y; a[6]=(_Float16)p1.z; a[7]=(_Float16)p1.w;
    a[8]=(_Float16)p2.x; a[9]=(_Float16)p2.y; a[10]=(_Float16)p2.z; a[11]=(_Float16)p2.w;
    a[12]=(_Float16)p3.x; a[13]=(_Float16)p3.y; a[14]=(_Float16)p3.z; a[15]=(_Float16)p3.w;

    // ---- B fragments from LDS: lanes 0-15 K=k0..k0+15, lanes 16-31 +16
    const int kb = k0 + hs * 16;
    v16h b0 = *(const v16h*)&wlds[(0 * 16 + l16) * Dn + kb];
    v16h b1 = *(const v16h*)&wlds[(1 * 16 + l16) * Dn + kb];
    v16h b2 = *(const v16h*)&wlds[(2 * 16 + l16) * Dn + kb];
    v16h b3 = *(const v16h*)&wlds[(3 * 16 + l16) * Dn + kb];

    acc0 = __builtin_amdgcn_wmma_f32_16x16x32_f16(false, a, false, b0, (short)0, acc0, false, false);
    acc1 = __builtin_amdgcn_wmma_f32_16x16x32_f16(false, a, false, b1, (short)0, acc1, false, false);
    acc2 = __builtin_amdgcn_wmma_f32_16x16x32_f16(false, a, false, b2, (short)0, acc2, false, false);
    acc3 = __builtin_amdgcn_wmma_f32_16x16x32_f16(false, a, false, b3, (short)0, acc3, false, false);
  }

  // ---- C/D layout: VGPR r -> M = r + 8*hs, N = l16 (per 16-col tile)
  const int mbase = mb * 128 + wave * 16 + hs * 8;
  const int ncol0 = ns * 64 + l16;
#pragma unroll
  for (int r = 0; r < 8; ++r) {
    float* crow = C + (size_t)(mbase + r) * N + ncol0;
    crow[0]  = acc0[r];
    crow[16] = acc1[r];
    crow[32] = acc2[r];
    crow[48] = acc3[r];
  }
}

// =====================================================================
// Zero-fill fp32 region
// =====================================================================
__global__ void zero_f32(float* __restrict__ p, long n) {
  long i = (long)blockIdx.x * blockDim.x + threadIdx.x;
  if (i < n) p[i] = 0.0f;
}

// =====================================================================
// Pass A: per (edge, head): logit = 10*tanh(MLP(q.k/4, w)),
// exp(logit) atomically accumulated into row/col denominators.
// MLP params staged in LDS (1032 floats) once per block.
// Thread t = e*8 + h.
// =====================================================================
__global__ __launch_bounds__(256)
void edge_logits(const float* __restrict__ qv1,   // (B*S, 256)
                 const float* __restrict__ kv2,   // (B*P, 256)
                 const int*   __restrict__ b_idx,
                 const int*   __restrict__ s_idx,
                 const int*   __restrict__ p_idx,
                 const float* __restrict__ w,
                 const float* __restrict__ msW1,  // (H,2,MS)
                 const float* __restrict__ msb1,  // (H,MS)
                 const float* __restrict__ msW2,  // (H,MS)
                 const float* __restrict__ msb2,  // (H)
                 float* __restrict__ logits,      // (E*H)
                 float* __restrict__ row_denom,   // (B*S*H)
                 float* __restrict__ col_denom) { // (B*P*H)
  __shared__ float sW1[Hn * 2 * MSn];   // 512
  __shared__ float sB1[Hn * MSn];       // 256
  __shared__ float sW2[Hn * MSn];       // 256
  __shared__ float sB2[Hn];             // 8

  const int tid = threadIdx.x;
  if (tid < Hn * 2 * MSn) sW1[tid] = msW1[tid];
  if (tid < Hn * MSn)     { sB1[tid] = msb1[tid]; sW2[tid] = msW2[tid]; }
  if (tid < Hn)           sB2[tid] = msb2[tid];
  __syncthreads();

  const long t = (long)blockIdx.x * blockDim.x + tid;
  if (t >= (long)En * Hn) return;
  const int e = (int)(t >> 3);
  const int h = (int)(t & 7);

  const int b  = b_idx[e];
  const long row = (long)b * Sn + s_idx[e];
  const long col = (long)b * Pn + p_idx[e];

  // q = qv1[row, sel=0, h, :], k = kv2[col, sel=0, h, :]
  const float4* __restrict__ q4 = (const float4*)(qv1 + row * 256 + h * 16);
  const float4* __restrict__ k4 = (const float4*)(kv2 + col * 256 + h * 16);
  float dot = 0.0f;
#pragma unroll
  for (int j = 0; j < 4; ++j) {
    float4 q = q4[j], k = k4[j];
    dot = fmaf(q.x, k.x, fmaf(q.y, k.y, fmaf(q.z, k.z, fmaf(q.w, k.w, dot))));
  }
  const float logit = dot * 0.25f;          // 1/sqrt(HD=16)
  const float we    = w[e];

  const float* __restrict__ w1a = sW1 + h * 2 * MSn;
  const float* __restrict__ w1b = w1a + MSn;
  const float* __restrict__ b1  = sB1 + h * MSn;
  const float* __restrict__ w2  = sW2 + h * MSn;
  float out = sB2[h];
#pragma unroll 8
  for (int m = 0; m < MSn; ++m) {
    float hid = fmaf(w1a[m], logit, fmaf(w1b[m], we, b1[m]));
    hid = fmaxf(hid, 0.0f);
    out = fmaf(hid, w2[m], out);
  }
  const float l = TANH_CLIP * tanhf(out);
  logits[t] = l;

  const float el = __expf(l);
  atomicAdd(&row_denom[row * Hn + h], el);   // global_atomic_add_f32 in L2
  atomicAdd(&col_denom[col * Hn + h], el);
}

// =====================================================================
// Pass B: attn = exp(l)/(denom+eps); scatter-add attn * values.
// row groups combine v2 (from kv2), col groups combine v1 (from qv1).
// =====================================================================
__global__ __launch_bounds__(256)
void edge_combine(const float* __restrict__ qv1,
                  const float* __restrict__ kv2,
                  const int*   __restrict__ b_idx,
                  const int*   __restrict__ s_idx,
                  const int*   __restrict__ p_idx,
                  const float* __restrict__ logits,
                  const float* __restrict__ row_denom,
                  const float* __restrict__ col_denom,
                  float* __restrict__ row_heads,   // (B*S, 128)
                  float* __restrict__ col_heads) { // (B*P, 128)
  const long t = (long)blockIdx.x * blockDim.x + threadIdx.x;
  if (t >= (long)En * Hn) return;
  const int e = (int)(t >> 3);
  const int h = (int)(t & 7);

  const int b  = b_idx[e];
  const long row = (long)b * Sn + s_idx[e];
  const long col = (long)b * Pn + p_idx[e];

  // v1 = qv1[row, sel=1, h, :], v2 = kv2[col, sel=1, h, :]
  const float4* __restrict__ v1 = (const float4*)(qv1 + row * 256 + 128 + h * 16);
  const float4* __restrict__ v2 = (const float4*)(kv2 + col * 256 + 128 + h * 16);
  __builtin_prefetch(v1, 0, 0);              // global_prefetch_b8: warm L2->L0
  __builtin_prefetch(v2, 0, 0);

  const float el = __expf(logits[t]);
  const float ar = el / (row_denom[row * Hn + h] + EPSc);
  const float ac = el / (col_denom[col * Hn + h] + EPSc);

  float* __restrict__ rh = row_heads + row * (Hn * HDn) + h * HDn;
  float* __restrict__ ch = col_heads + col * (Hn * HDn) + h * HDn;
#pragma unroll
  for (int j = 0; j < 4; ++j) {
    float4 a = v2[j];     // row side combines v2
    float4 c = v1[j];     // col side combines v1
    atomicAdd(&rh[j * 4 + 0], ar * a.x);
    atomicAdd(&rh[j * 4 + 1], ar * a.y);
    atomicAdd(&rh[j * 4 + 2], ar * a.z);
    atomicAdd(&rh[j * 4 + 3], ar * a.w);
    atomicAdd(&ch[j * 4 + 0], ac * c.x);
    atomicAdd(&ch[j * 4 + 1], ac * c.y);
    atomicAdd(&ch[j * 4 + 2], ac * c.z);
    atomicAdd(&ch[j * 4 + 3], ac * c.w);
  }
}

// =====================================================================
// Launch
// =====================================================================
extern "C" void kernel_launch(void* const* d_in, const int* in_sizes, int n_in,
                              void* d_out, int out_size, void* d_ws, size_t ws_size,
                              hipStream_t stream) {
  const float* x1    = (const float*)d_in[0];
  const float* x2    = (const float*)d_in[1];
  const int*   b_idx = (const int*)  d_in[2];
  const int*   s_idx = (const int*)  d_in[3];
  const int*   p_idx = (const int*)  d_in[4];
  const float* w     = (const float*)d_in[5];
  const float* Wqv1  = (const float*)d_in[6];
  const float* Wkv2  = (const float*)d_in[7];
  const float* Wo1   = (const float*)d_in[8];
  const float* Wo2   = (const float*)d_in[9];
  const float* msW1  = (const float*)d_in[10];
  const float* msb1  = (const float*)d_in[11];
  const float* msW2  = (const float*)d_in[12];
  const float* msb2  = (const float*)d_in[13];
  float* out = (float*)d_out;                  // h1 (B*S*D) ‖ h2 (B*P*D)

  // ---- workspace layout (floats) ----
  float* ws = (float*)d_ws;
  const long nRows = (long)Bn * Sn;            // 16384
  const long nCols = (long)Bn * Pn;            // 16384
  float* qv1       = ws;                        // nRows*256
  float* kv2       = qv1 + nRows * 256;         // nCols*256
  float* logits    = kv2 + nCols * 256;         // E*H
  float* row_denom = logits + (long)En * Hn;    // nRows*H
  float* col_denom = row_denom + nRows * Hn;    // nCols*H
  float* row_heads = col_denom + nCols * Hn;    // nRows*128
  float* col_heads = row_heads + nRows * (Hn * HDn); // nCols*128

  // ---- 0) zero the accumulators (denoms + heads are contiguous) ----
  {
    long nz = (nRows + nCols) * Hn + (nRows + nCols) * (Hn * HDn);
    int blocks = (int)((nz + 255) / 256);
    zero_f32<<<blocks, 256, 0, stream>>>(row_denom, nz);
  }

  // ---- 1) projections via WMMA: qv1 = x1@Wqv1, kv2 = x2@Wkv2 ----
  {
    // block tile 128x64: (16384/128) * (256/64) = 512 blocks
    int blocks = (int)(nRows / 128) * (256 / 64);
    gemm_wmma_f16<<<blocks, 256, 0, stream>>>(x1, Wqv1, qv1, (int)nRows, 256);
    gemm_wmma_f16<<<blocks, 256, 0, stream>>>(x2, Wkv2, kv2, (int)nCols, 256);
  }

  // ---- 2) edge logits + softmax denominators ----
  {
    long nt = (long)En * Hn;                   // 4M threads, exact multiple of 256
    int blocks = (int)((nt + 255) / 256);
    edge_logits<<<blocks, 256, 0, stream>>>(qv1, kv2, b_idx, s_idx, p_idx, w,
                                            msW1, msb1, msW2, msb2,
                                            logits, row_denom, col_denom);
  }

  // ---- 3) softmax-weighted combine (scatter) ----
  {
    long nt = (long)En * Hn;
    int blocks = (int)((nt + 255) / 256);
    edge_combine<<<blocks, 256, 0, stream>>>(qv1, kv2, b_idx, s_idx, p_idx,
                                             logits, row_denom, col_denom,
                                             row_heads, col_heads);
  }

  // ---- 4) output projections via WMMA ----
  {
    // block tile 128x64: (16384/128) * (128/64) = 256 blocks
    int blocks = (int)(nRows / 128) * (Dn / 64);
    gemm_wmma_f16<<<blocks, 256, 0, stream>>>(row_heads, Wo1, out, (int)nRows, Dn);
    gemm_wmma_f16<<<blocks, 256, 0, stream>>>(col_heads, Wo2, out + nRows * Dn,
                                              (int)nCols, Dn);
  }
}
